// ConvNextV2Block_30958124269767
// MI455X (gfx1250) — matrix operand order repaired and checked by reading, not compiled, not run on hardware
//
#include <hip/hip_runtime.h>
#include <cstdint>
#include <cstddef>

typedef __bf16 bf16_t;
typedef bf16_t v16bf __attribute__((ext_vector_type(16)));
typedef float  v8f   __attribute__((ext_vector_type(8)));

#define BB   2
#define CC   96
#define DDim 24
#define HHd  56
#define WWd  56
#define EE   384
#define HWP  3136      // 56*56
#define DHW  75264     // 24*56*56
#define NPOS 150528    // BB*DHW

union Frag { uint4 q[2]; v16bf v; };

__device__ __forceinline__ unsigned short f2bf(float f) {
    unsigned u = __float_as_uint(f);
    u += 0x7FFFu + ((u >> 16) & 1u);
    return (unsigned short)(u >> 16);
}
__device__ __forceinline__ float bf2f(unsigned short h) {
    return __uint_as_float(((unsigned)h) << 16);
}

// Async global->LDS 16B copy (CDNA5 GLOBAL_LOAD_ASYNC_TO_LDS_B128, ASYNCcnt).
// Low 32 bits of a flat LDS pointer are the wave-relative LDS address.
__device__ __forceinline__ void async_copy_b128(const void* g, void* l) {
    unsigned loff = (unsigned)(uintptr_t)l;
    asm volatile("global_load_async_to_lds_b128 %0, %1, off"
                 :: "v"(loff), "v"(g) : "memory");
}
__device__ __forceinline__ void async_wait0() {
    asm volatile("s_wait_asynccnt 0x0" ::: "memory");
}

// ---------------------------------------------------------------------------
// Weight prep: fp32 -> bf16, pre-swizzled into WMMA B-fragment order.
// Layout: [kstep][ntile][lane][16], lane = column (n = nt*16 + (lane&15)),
// contiguous K per lane: k = ks*32 + (lane>>4)*16 + i  (same k order as the A loads).
// ---------------------------------------------------------------------------
__global__ void k_prep_weights(const float* __restrict__ w1,
                               const float* __restrict__ w2,
                               unsigned short* __restrict__ w1s,
                               unsigned short* __restrict__ w2s) {
    int t = blockIdx.x * 256 + threadIdx.x;
    if (t < CC * EE) {
        int i  = t & 15;
        int l  = (t >> 4) & 31;
        int nt = (t >> 9) % 24;
        int ks = (t >> 9) / 24;                       // 0..2
        int k  = ks * 32 + ((l >> 4) << 4) + i;       // 0..95
        int n  = nt * 16 + (l & 15);                  // 0..383
        w1s[t] = f2bf(w1[k * EE + n]);
    } else {
        int t2 = t - CC * EE;
        if (t2 < EE * CC) {
            int i  = t2 & 15;
            int l  = (t2 >> 4) & 31;
            int nt = (t2 >> 9) % 6;
            int ks = (t2 >> 9) / 6;                   // 0..11
            int k  = ks * 32 + ((l >> 4) << 4) + i;   // 0..383
            int n  = nt * 16 + (l & 15);              // 0..95
            w2s[t2] = f2bf(w2[k * CC + n]);
        }
    }
}

// cvec[j] = b2[j] + sum_e beta[e]*w2[e,j]; zero GRN accumulator.
__global__ void k_prep_small(const float* __restrict__ w2,
                             const float* __restrict__ b2,
                             const float* __restrict__ beta,
                             float* __restrict__ gsq,
                             float* __restrict__ cvec) {
    int t = threadIdx.x;
    if (t < BB * EE) gsq[t] = 0.f;
    if (t < CC) {
        float s = b2[t];
        for (int e = 0; e < EE; ++e) s += beta[e] * w2[e * CC + t];
        cvec[t] = s;
    }
}

// ---------------------------------------------------------------------------
// Depthwise 7x7x7 conv + bias, channels-first in/out, LDS halo tile.
// grid(14, 24, 192) = (HWtile, d, b*96+c), block 256 = 8x32 HW tile.
// ---------------------------------------------------------------------------
__global__ void __launch_bounds__(256)
k_dwconv(const float* __restrict__ x, const float* __restrict__ wk,
         const float* __restrict__ bias, float* __restrict__ ydw) {
    __shared__ float tile[7 * 14 * 38];
    const int z  = blockIdx.z;            // b*96 + c
    const int c  = z % CC;
    const int d  = blockIdx.y;
    const int h0 = (blockIdx.x >> 1) * 8;
    const int w0 = (blockIdx.x & 1) * 32;
    const int tid = threadIdx.x;
    const float* xs = x + (size_t)z * DHW;

    for (int li = tid; li < 7 * 14 * 38; li += 256) {
        int kd = li / (14 * 38);
        int r  = li % (14 * 38);
        int hh = r / 38, ww = r % 38;
        int gd = d + kd - 3, gh = h0 + hh - 3, gw = w0 + ww - 3;
        float v = 0.f;
        if (gd >= 0 && gd < DDim && gh >= 0 && gh < HHd && gw >= 0 && gw < WWd)
            v = xs[gd * HWP + gh * WWd + gw];
        tile[li] = v;
    }
    __syncthreads();

    int hh = tid >> 5, ww = tid & 31;
    int h = h0 + hh, w = w0 + ww;
    if (h < HHd && w < WWd) {
        const float* kc = wk + c * 343;   // block-uniform -> scalar loads
        float acc = 0.f;
        for (int kd = 0; kd < 7; ++kd)
            for (int kh = 0; kh < 7; ++kh) {
                const float* trow = &tile[(kd * 14 + hh + kh) * 38 + ww];
                const float* krow = &kc[(kd * 7 + kh) * 7];
                #pragma unroll
                for (int kw = 0; kw < 7; ++kw)
                    acc = fmaf(trow[kw], krow[kw], acc);
            }
        ydw[(size_t)z * DHW + d * HWP + h * WWd + w] = acc + bias[c];
    }
}

// ---------------------------------------------------------------------------
// LayerNorm over C=96, lane-per-position; writes bf16 A matrix [pos, 96].
// grid 588, block 256 (8 waves x 32 positions).
// ---------------------------------------------------------------------------
__global__ void __launch_bounds__(256)
k_ln(const float* __restrict__ ydw, const float* __restrict__ lw,
     const float* __restrict__ lb, unsigned short* __restrict__ A) {
    int wv = threadIdx.x >> 5, lane = threadIdx.x & 31;
    int pos0 = (blockIdx.x * 8 + wv) * 32;
    int pos  = pos0 + lane;
    int b    = pos0 / DHW;                // DHW % 32 == 0 -> uniform per wave
    int sp   = pos - b * DHW;
    const float* src = ydw + (size_t)b * CC * DHW + sp;

    float sum = 0.f, sq = 0.f;
    #pragma unroll
    for (int cidx = 0; cidx < CC; ++cidx) {
        float v = src[(size_t)cidx * DHW];
        sum += v; sq += v * v;
    }
    float mean = sum * (1.f / 96.f);
    float var  = sq * (1.f / 96.f) - mean * mean;
    float rs   = rsqrtf(var + 1e-6f);

    unsigned short* dst = A + (size_t)pos * CC;
    #pragma unroll
    for (int cb = 0; cb < CC; cb += 8) {
        unsigned short o[8];
        #pragma unroll
        for (int j = 0; j < 8; ++j) {
            int cidx = cb + j;
            float v = src[(size_t)cidx * DHW];
            v = (v - mean) * rs * lw[cidx] + lb[cidx];
            o[j] = f2bf(v);
        }
        uint4 pk;
        pk.x = (unsigned)o[0] | ((unsigned)o[1] << 16);
        pk.y = (unsigned)o[2] | ((unsigned)o[3] << 16);
        pk.z = (unsigned)o[4] | ((unsigned)o[5] << 16);
        pk.w = (unsigned)o[6] | ((unsigned)o[7] << 16);
        *(uint4*)(dst + cb) = pk;
    }
}

// ---------------------------------------------------------------------------
// GEMM1: [NPOS,96]x[96,384] bf16 WMMA, + b1, exact GELU, bf16 act out,
// + per-(batch,channel) sum-of-squares for GRN (LDS ds_add -> global atomic).
// grid 3528 blocks x 8 waves; wave tile 32x64 (2 M x 4 N fragments).
// B chunks staged global->LDS via async DMA, double-buffered.
// ---------------------------------------------------------------------------
__global__ void __launch_bounds__(256)
k_gemm1(const unsigned short* __restrict__ A,
        const unsigned short* __restrict__ w1s,
        const float* __restrict__ b1, float* __restrict__ gsq,
        unsigned short* __restrict__ act) {
    __shared__ unsigned short Bs[2][24 * 32 * 16];   // 2 x 24 KB ping-pong
    __shared__ float colsq[EE];
    const int tid = threadIdx.x;
    const int wv = tid >> 5, lane = tid & 31;
    const int waveId = blockIdx.x * 8 + wv;
    const int mwave = waveId / 6;
    const int nwave = waveId % 6;
    const int pos0 = mwave * 32;
    const int n0 = nwave * 64;
    const int lhalf = lane >> 4, lcol = lane & 15;

    for (int t = tid; t < EE; t += 256) colsq[t] = 0.f;

    v8f zero = {0.f, 0.f, 0.f, 0.f, 0.f, 0.f, 0.f, 0.f};
    v8f acc[2][4];
    #pragma unroll
    for (int i = 0; i < 2; ++i)
        #pragma unroll
        for (int j = 0; j < 4; ++j) acc[i][j] = zero;

    // async-issue one 24KB B chunk (6 x b128 per thread)
    auto issueB = [&](int ks, int buf) {
        const unsigned short* src = w1s + ks * 24 * 32 * 16;
        #pragma unroll
        for (int i = 0; i < 6; ++i) {
            int id = tid + i * 256;
            async_copy_b128(src + id * 8, &Bs[buf][id * 8]);
        }
    };

    issueB(0, 0);
    for (int ks = 0; ks < 3; ++ks) {
        const int buf = ks & 1;
        async_wait0();
        __syncthreads();
        if (ks + 1 < 3) issueB(ks + 1, buf ^ 1);

        Frag a[2];
        #pragma unroll
        for (int i = 0; i < 2; ++i) {
            const unsigned short* ap =
                A + (size_t)(pos0 + i * 16 + lcol) * CC + ks * 32;
            a[i].q[0] = *(const uint4*)(ap + lhalf * 16);       // K 0-7 / 16-23
            a[i].q[1] = *(const uint4*)(ap + lhalf * 16 + 8);   // K 8-15 / 24-31
        }
        #pragma unroll
        for (int j = 0; j < 4; ++j) {
            Frag bfr;
            const unsigned short* bp =
                &Bs[buf][((nwave * 4 + j) * 32 + lane) * 16];
            bfr.q[0] = *(const uint4*)bp;
            bfr.q[1] = *(const uint4*)(bp + 8);
            #pragma unroll
            for (int i = 0; i < 2; ++i)
                acc[i][j] = __builtin_amdgcn_wmma_f32_16x16x32_bf16(
                    false, a[i].v, false, bfr.v, (short)0, acc[i][j],
                    false, false);
        }
        __syncthreads();   // all waves done reading buf before it is refilled
    }

    // epilogue: bias + exact GELU + bf16 store + column sum of squares
    #pragma unroll
    for (int j = 0; j < 4; ++j) {
        int col = n0 + j * 16 + lcol;
        float bias = b1[col];
        float sq = 0.f;
        #pragma unroll
        for (int i = 0; i < 2; ++i) {
            int rbase = pos0 + i * 16 + lhalf * 8;
            #pragma unroll
            for (int r = 0; r < 8; ++r) {
                float v = acc[i][j][r] + bias;
                float g = 0.5f * v * (1.f + erff(v * 0.70710678118654752f));
                sq += g * g;
                act[(size_t)(rbase + r) * EE + col] = f2bf(g);
            }
        }
        sq += __shfl_xor(sq, 16, 32);
        if (lane < 16) atomicAdd(&colsq[col], sq);
    }
    __syncthreads();
    // batch is uniform per block (DHW*6/32 waves divisible by 8)
    const int bBlk = (((blockIdx.x * 8) / 6) * 32) / DHW;
    for (int t = tid; t < EE; t += 256)
        atomicAdd(&gsq[bBlk * EE + t], colsq[t]);
}

// GRN scale: s[b,e] = 1 + gamma[e] * gx / (mean_e(gx) + 1e-6)
__global__ void k_grn(const float* __restrict__ gsq,
                      const float* __restrict__ gamma,
                      float* __restrict__ sfac) {
    __shared__ float sm[EE];
    int b = blockIdx.x, t = threadIdx.x;
    float g = sqrtf(gsq[b * EE + t]);
    sm[t] = g;
    __syncthreads();
    if (t < 128) sm[t] += sm[t + 256];
    __syncthreads();
    for (int s = 128; s > 0; s >>= 1) {
        if (t < s) sm[t] += sm[t + s];
        __syncthreads();
    }
    float mean = sm[0] * (1.f / 384.f);
    sfac[b * EE + t] = 1.f + gamma[t] * (g / (mean + 1e-6f));
}

// act[pos,e] *= s[b,e]  (8 bf16 per thread)
__global__ void __launch_bounds__(256)
k_scale(unsigned short* __restrict__ act, const float* __restrict__ sfac) {
    size_t t = (size_t)blockIdx.x * 256 + threadIdx.x;
    size_t f = t * 8;
    int e0  = (int)(f % EE);
    int pos = (int)(f / EE);
    int b   = pos / DHW;
    const float* s = sfac + b * EE + e0;
    uint4 v = *(uint4*)(act + f);
    unsigned u[4] = {v.x, v.y, v.z, v.w};
    #pragma unroll
    for (int i = 0; i < 4; ++i) {
        float lo = bf2f((unsigned short)(u[i] & 0xFFFF)) * s[2 * i];
        float hi = bf2f((unsigned short)(u[i] >> 16))    * s[2 * i + 1];
        u[i] = (unsigned)f2bf(lo) | ((unsigned)f2bf(hi) << 16);
    }
    v.x = u[0]; v.y = u[1]; v.z = u[2]; v.w = u[3];
    *(uint4*)(act + f) = v;
}

// ---------------------------------------------------------------------------
// GEMM2: [NPOS,384]x[384,96] bf16 WMMA + cvec + residual, channels-first out.
// grid 588 blocks x 8 waves; wave tile 32x96 (2 M x 6 N fragments).
// B chunks staged global->LDS via async DMA, double-buffered (12 k-steps).
// ---------------------------------------------------------------------------
__global__ void __launch_bounds__(256)
k_gemm2(const unsigned short* __restrict__ act,
        const unsigned short* __restrict__ w2s,
        const float* __restrict__ cvec, const float* __restrict__ x,
        float* __restrict__ out) {
    __shared__ unsigned short Bs[2][6 * 32 * 16];    // 2 x 6 KB ping-pong
    const int tid = threadIdx.x, wv = tid >> 5, lane = tid & 31;
    const int m0 = (blockIdx.x * 8 + wv) * 32;
    const int lhalf = lane >> 4, lcol = lane & 15;

    v8f zero = {0.f, 0.f, 0.f, 0.f, 0.f, 0.f, 0.f, 0.f};
    v8f acc[2][6];
    #pragma unroll
    for (int i = 0; i < 2; ++i)
        #pragma unroll
        for (int j = 0; j < 6; ++j) acc[i][j] = zero;

    auto issueB = [&](int ks, int buf) {
        const unsigned short* src = w2s + ks * 6 * 32 * 16;
        #pragma unroll
        for (int i = 0; i < 2; ++i) {
            int id = tid + i * 256;
            if (id < 384)
                async_copy_b128(src + id * 8, &Bs[buf][id * 8]);
        }
    };

    issueB(0, 0);
    for (int ks = 0; ks < 12; ++ks) {
        const int buf = ks & 1;
        async_wait0();
        __syncthreads();
        if (ks + 1 < 12) issueB(ks + 1, buf ^ 1);

        Frag a[2];
        #pragma unroll
        for (int i = 0; i < 2; ++i) {
            const unsigned short* ap =
                act + (size_t)(m0 + i * 16 + lcol) * EE + ks * 32;
            a[i].q[0] = *(const uint4*)(ap + lhalf * 16);
            a[i].q[1] = *(const uint4*)(ap + lhalf * 16 + 8);
            if (ks + 1 < 12) __builtin_prefetch(ap + 32, 0, 1);
        }
        #pragma unroll
        for (int j = 0; j < 6; ++j) {
            Frag bfr;
            const unsigned short* bp = &Bs[buf][(j * 32 + lane) * 16];
            bfr.q[0] = *(const uint4*)bp;
            bfr.q[1] = *(const uint4*)(bp + 8);
            #pragma unroll
            for (int i = 0; i < 2; ++i)
                acc[i][j] = __builtin_amdgcn_wmma_f32_16x16x32_bf16(
                    false, a[i].v, false, bfr.v, (short)0, acc[i][j],
                    false, false);
        }
        __syncthreads();   // all waves done reading buf before it is refilled
    }

    const int b = m0 / DHW;
    const int sp0w = m0 - b * DHW;
    #pragma unroll
    for (int j = 0; j < 6; ++j) {
        int ch = j * 16 + lcol;
        float cv = cvec[ch];
        const float* xr = x   + (size_t)(b * CC + ch) * DHW;
        float*      orow = out + (size_t)(b * CC + ch) * DHW;
        #pragma unroll
        for (int i = 0; i < 2; ++i) {
            int sp = sp0w + i * 16 + lhalf * 8;
            float4 x0 = *(const float4*)(xr + sp);
            float4 x1 = *(const float4*)(xr + sp + 4);
            float4 o0, o1;
            o0.x = x0.x + acc[i][j][0] + cv;
            o0.y = x0.y + acc[i][j][1] + cv;
            o0.z = x0.z + acc[i][j][2] + cv;
            o0.w = x0.w + acc[i][j][3] + cv;
            o1.x = x1.x + acc[i][j][4] + cv;
            o1.y = x1.y + acc[i][j][5] + cv;
            o1.z = x1.z + acc[i][j][6] + cv;
            o1.w = x1.w + acc[i][j][7] + cv;
            *(float4*)(orow + sp)     = o0;
            *(float4*)(orow + sp + 4) = o1;
        }
    }
}

// ---------------------------------------------------------------------------
extern "C" void kernel_launch(void* const* d_in, const int* in_sizes, int n_in,
                              void* d_out, int out_size, void* d_ws,
                              size_t ws_size, hipStream_t stream) {
    const float* x    = (const float*)d_in[0];
    const float* dwk  = (const float*)d_in[1];
    const float* dwb  = (const float*)d_in[2];
    const float* lnw  = (const float*)d_in[3];
    const float* lnb  = (const float*)d_in[4];
    const float* w1   = (const float*)d_in[5];
    const float* b1   = (const float*)d_in[6];
    const float* gam  = (const float*)d_in[7];
    const float* bet  = (const float*)d_in[8];
    const float* w2   = (const float*)d_in[9];
    const float* b2   = (const float*)d_in[10];
    float* out = (float*)d_out;

    char* ws = (char*)d_ws;
    float*          ydw = (float*)ws;                             //  57,802,752 B
    unsigned short* Abf = (unsigned short*)(ws + 57802752);       //  28,901,376 B
    unsigned short* act = (unsigned short*)(ws + 86704128);       // 115,605,504 B
    unsigned short* w1s = (unsigned short*)(ws + 202309632);      //      73,728 B
    unsigned short* w2s = (unsigned short*)(ws + 202383360);      //      73,728 B
    float*          gsq = (float*)(ws + 202457088);               //       3,072 B
    float*          sfc = (float*)(ws + 202460160);               //       3,072 B
    float*          cvc = (float*)(ws + 202463232);               //         384 B

    k_prep_weights<<<288, 256, 0, stream>>>(w1, w2, w1s, w2s);
    k_prep_small<<<1, 768, 0, stream>>>(w2, b2, bet, gsq, cvc);
    k_dwconv<<<dim3(14, 24, 192), 256, 0, stream>>>(x, dwk, dwb, ydw);
    k_ln<<<588, 256, 0, stream>>>(ydw, lnw, lnb, Abf);
    k_gemm1<<<3528, 256, 0, stream>>>(Abf, w1s, b1, gsq, act);
    k_grn<<<2, 384, 0, stream>>>(gsq, gam, sfc);
    k_scale<<<28224, 256, 0, stream>>>(act, sfc);
    k_gemm2<<<588, 256, 0, stream>>>(act, w2s, cvc, x, out);
}
